// NysAttention_8358006358107
// MI455X (gfx1250) — compile-verified, matrix operand order checked by reading, not run on hardware
//
#include <hip/hip_runtime.h>
#include <hip/hip_bf16.h>

// ---------------------------------------------------------------------------
// Nystromformer attention for MI455X (gfx1250, wave32, WMMA)
// B=8, N=4096, C=768, H=12, M=64, D=64
// Kernels 1,4,6,7 use v_wmma_f32_16x16x32_bf16 (f32 accumulate); kernels 3,5 use fp32 VALU.
// ---------------------------------------------------------------------------

#define BB 8
#define NN 4096
#define CC 768
#define HH 12
#define MM 64
#define DD 64

typedef __bf16 v16bf __attribute__((ext_vector_type(16)));
typedef float  v8f   __attribute__((ext_vector_type(8)));

static __device__ __forceinline__ __bf16 f2bf(float f) {
    unsigned u = __builtin_bit_cast(unsigned, f);
    u += 0x7FFFu + ((u >> 16) & 1u);            // round-to-nearest-even
    unsigned short h = (unsigned short)(u >> 16);
    return __builtin_bit_cast(__bf16, h);
}
static __device__ __forceinline__ float bf2f(__bf16 b) {
    unsigned short h = __builtin_bit_cast(unsigned short, b);
    unsigned u = ((unsigned)h) << 16;
    return __builtin_bit_cast(float, u);
}

// Load a 16x32 bf16 operand fragment from LDS.
// Logical layout of the LDS array: [row][k], row stride = stride halves
// (k contiguous).  Lane L holds row (passed in, caller adds lane&15);
// half h = L>>4 selects K runs [8h,8h+8) and [16+8h,24+8h) per ISA layout.
static __device__ __forceinline__ v16bf ldfrag(const __bf16* s, int row,
                                               int stride, int kbase, int lane) {
    const unsigned* su = (const unsigned*)s;
    int h = lane >> 4;
    int base = row * stride + kbase + 8 * h;    // in halves; even by construction
    union { v16bf v; unsigned u[8]; } f;
#pragma unroll
    for (int j = 0; j < 4; ++j) {
        f.u[j]     = su[(base + 2 * j) >> 1];
        f.u[j + 4] = su[(base + 16 + 2 * j) >> 1];
    }
    return f.v;
}

static __device__ __forceinline__ v8f wmma_bf16(v16bf a, v16bf b, v8f c) {
    return __builtin_amdgcn_wmma_f32_16x16x32_bf16(
        /*neg_a=*/false, a, /*neg_b=*/false, b,
        /*c_mod=*/(short)0, c, /*reuse_a=*/false, /*reuse_b=*/false);
}

// ---------------------------------------------------------------------------
// Kernel 1: qkv = x @ w_qkv^T ; scatter into Q(/8), K, V  [B,H,N,D] bf16
// grid (256, 36), block 256 (8 waves). Block tile 128 rows x 64 cols.
// ---------------------------------------------------------------------------
__global__ __launch_bounds__(256) void k_gemm_qkv(
    const float* __restrict__ x, const float* __restrict__ w,
    __bf16* __restrict__ Qb, __bf16* __restrict__ Kb, __bf16* __restrict__ Vb) {
    __shared__ __align__(16) __bf16 As[128 * 32];
    __shared__ __align__(16) __bf16 Bs[64 * 32];
    const int tid  = threadIdx.x;
    const int lane = tid & 31;
    const int w8   = tid >> 5;
    const int wr   = w8 >> 1;   // 0..3  (32-row strip)
    const int wc   = w8 & 1;    // 0..1  (32-col strip)
    const int r0   = blockIdx.x * 128;
    const int c0   = blockIdx.y * 64;
    const int m15  = lane & 15;

    v8f acc[2][2];
#pragma unroll
    for (int i = 0; i < 2; ++i)
#pragma unroll
        for (int j = 0; j < 2; ++j) acc[i][j] = (v8f){0,0,0,0,0,0,0,0};

    for (int kt = 0; kt < CC; kt += 32) {
        if (kt + 32 < CC)
            __builtin_prefetch(&x[(size_t)(r0 + (tid >> 1)) * CC + kt + 32], 0, 1);
#pragma unroll
        for (int i = 0; i < 16; ++i) {
            int e = i * 256 + tid, row = e >> 5, col = e & 31;
            As[e] = f2bf(x[(size_t)(r0 + row) * CC + kt + col]);
        }
#pragma unroll
        for (int i = 0; i < 8; ++i) {
            int e = i * 256 + tid, row = e >> 5, col = e & 31;
            Bs[e] = f2bf(w[(size_t)(c0 + row) * CC + kt + col]);
        }
        __syncthreads();
#pragma unroll
        for (int i = 0; i < 2; ++i) {
            v16bf a = ldfrag(As, wr * 32 + i * 16 + m15, 32, 0, lane);
#pragma unroll
            for (int j = 0; j < 2; ++j) {
                v16bf b = ldfrag(Bs, wc * 32 + j * 16 + m15, 32, 0, lane);
                acc[i][j] = wmma_bf16(a, b, acc[i][j]);
            }
        }
        __syncthreads();
    }

    // Epilogue: block col range [c0,c0+64) lies inside one (t,h) pair.
    const int t = c0 / CC;
    const int h = (c0 % CC) / DD;
    __bf16* dst = (t == 0) ? Qb : ((t == 1) ? Kb : Vb);
    const float scl = (t == 0) ? 0.125f : 1.0f;   // 1/sqrt(D)
    const int lh = lane >> 4;
#pragma unroll
    for (int i = 0; i < 2; ++i)
#pragma unroll
        for (int j = 0; j < 2; ++j)
#pragma unroll
            for (int r = 0; r < 8; ++r) {
                int m  = wr * 32 + i * 16 + r + 8 * lh;
                int gr = r0 + m;
                int b  = gr >> 12, n = gr & (NN - 1);
                int d  = wc * 32 + j * 16 + m15;
                dst[(((size_t)b * HH + h) * NN + n) * DD + d] =
                    f2bf(acc[i][j][r] * scl);
            }
}

// ---------------------------------------------------------------------------
// Kernel 2: landmark pooling (segment means of 64) -> Ql, Kl [B,H,M,D] bf16
// grid B*H*M, block 64.
// ---------------------------------------------------------------------------
__global__ __launch_bounds__(64) void k_landmark(
    const __bf16* __restrict__ Qb, const __bf16* __restrict__ Kb,
    __bf16* __restrict__ Qlb, __bf16* __restrict__ Klb) {
    const int id = blockIdx.x;
    const int m  = id & (MM - 1);
    const int bh = id >> 6;
    const int d  = threadIdx.x;
    size_t base = ((size_t)bh * NN + (size_t)m * 64) * DD + d;
    float sq = 0.f, sk = 0.f;
#pragma unroll 4
    for (int s = 0; s < 64; ++s) {
        sq += bf2f(Qb[base + (size_t)s * DD]);
        sk += bf2f(Kb[base + (size_t)s * DD]);
    }
    size_t o = ((size_t)bh * MM + m) * DD + d;
    Qlb[o] = f2bf(sq * (1.f / 64.f));
    Klb[o] = f2bf(sk * (1.f / 64.f));
}

// ---------------------------------------------------------------------------
// Kernel 3: kernel2 = softmax(Ql @ Kl^T); Newton-Schulz pinv (6 iters).
// One block per (b,h), 256 threads, all in LDS (fp32).
// ---------------------------------------------------------------------------
__device__ __forceinline__ void mm64_bt(float* O, const float* A, const float* Bm) {
    // O = A @ Bm^T
#pragma unroll
    for (int i = 0; i < 16; ++i) {
        int e = i * 256 + threadIdx.x, r = e >> 6, c = e & 63;
        float a = 0.f;
#pragma unroll
        for (int k = 0; k < 64; ++k) a += A[r * 64 + k] * Bm[c * 64 + k];
        O[e] = a;
    }
}
__device__ __forceinline__ void mm64_k2(float* O, const __bf16* A, const float* Bm) {
    // O = A(bf16) @ Bm
#pragma unroll
    for (int i = 0; i < 16; ++i) {
        int e = i * 256 + threadIdx.x, r = e >> 6, c = e & 63;
        float a = 0.f;
#pragma unroll
        for (int k = 0; k < 64; ++k) a += bf2f(A[r * 64 + k]) * Bm[k * 64 + c];
        O[e] = a;
    }
}
__device__ __forceinline__ void mm64_bt7(float* O, const float* KVm) {
    // O = KV @ (7I - KV)
#pragma unroll
    for (int i = 0; i < 16; ++i) {
        int e = i * 256 + threadIdx.x, r = e >> 6, c = e & 63;
        float a = 0.f;
#pragma unroll
        for (int k = 0; k < 64; ++k)
            a += KVm[r * 64 + k] * (((k == c) ? 7.f : 0.f) - KVm[k * 64 + c]);
        O[e] = a;
    }
}
__device__ __forceinline__ void mm64_inplaceA(float* A, const float* Bm) {
    // A = A @ Bm (in place on A); 4 threads per row.
    const int r  = threadIdx.x >> 2;
    const int cg = (threadIdx.x & 3) * 16;
    float areg[64];
#pragma unroll
    for (int k = 0; k < 64; ++k) areg[k] = A[r * 64 + k];
    __syncthreads();
    float o[16];
#pragma unroll
    for (int c = 0; c < 16; ++c) {
        float a = 0.f;
#pragma unroll
        for (int k = 0; k < 64; ++k) a += areg[k] * Bm[k * 64 + cg + c];
        o[c] = a;
    }
#pragma unroll
    for (int c = 0; c < 16; ++c) A[r * 64 + cg + c] = o[c];
}

__global__ __launch_bounds__(256) void k_inv(
    const __bf16* __restrict__ Qlb, const __bf16* __restrict__ Klb,
    __bf16* __restrict__ K2INVb) {
    __shared__ __bf16 K2b[64 * 64];
    __shared__ float  Vv[64 * 64];
    __shared__ float  KV[64 * 64];
    __shared__ float  T1[64 * 64];
    __shared__ float  red[64];
    const int tid = threadIdx.x;
    const size_t bh = blockIdx.x;
    const size_t gb = bh * (MM * DD);

#pragma unroll
    for (int i = 0; i < 16; ++i) {
        int e = i * 256 + tid;
        KV[e] = bf2f(Qlb[gb + e]);
        T1[e] = bf2f(Klb[gb + e]);
    }
    __syncthreads();
    mm64_bt(Vv, KV, T1);                       // logits2 = Ql @ Kl^T
    __syncthreads();
    if (tid < 64) {                            // softmax rows (with max-sub)
        float mx = -1e30f;
        for (int j = 0; j < 64; ++j) mx = fmaxf(mx, Vv[tid * 64 + j]);
        float s = 0.f;
        for (int j = 0; j < 64; ++j) {
            float ev = __expf(Vv[tid * 64 + j] - mx);
            Vv[tid * 64 + j] = ev; s += ev;
        }
        float inv = 1.f / s;
        for (int j = 0; j < 64; ++j) Vv[tid * 64 + j] *= inv;
    }
    __syncthreads();
#pragma unroll
    for (int i = 0; i < 16; ++i) { int e = i * 256 + tid; K2b[e] = f2bf(Vv[e]); }
    __syncthreads();
    if (tid < 64) {                            // column sums
        float s = 0.f;
        for (int i = 0; i < 64; ++i) s += Vv[i * 64 + tid];
        red[tid] = s;
    }
    __syncthreads();
    if (tid == 0) {
        float mx = red[0];
        for (int j = 1; j < 64; ++j) mx = fmaxf(mx, red[j]);
        red[0] = mx;
    }
    __syncthreads();
    const float inv_scale = 1.f / red[0];
    __syncthreads();
#pragma unroll
    for (int i = 0; i < 16; ++i) {             // V0 = K2^T / scale
        int e = i * 256 + tid, r = e >> 6, c = e & 63;
        Vv[e] = bf2f(K2b[c * 64 + r]) * inv_scale;
    }
    __syncthreads();
    for (int it = 0; it < 6; ++it) {
        mm64_k2(KV, K2b, Vv);                  // KV = K @ V
        __syncthreads();
        mm64_bt7(T1, KV);                      // T1 = KV @ (7I - KV)
        __syncthreads();
#pragma unroll
        for (int i = 0; i < 16; ++i) {         // T1 = 15I - T1
            int e = i * 256 + tid, r = e >> 6, c = e & 63;
            T1[e] = ((r == c) ? 15.f : 0.f) - T1[e];
        }
        __syncthreads();
        mm64_inplaceA(KV, T1);                 // KV = KV @ T1
        __syncthreads();
#pragma unroll
        for (int i = 0; i < 16; ++i) {         // KV = 13I - KV
            int e = i * 256 + tid, r = e >> 6, c = e & 63;
            KV[e] = ((r == c) ? 13.f : 0.f) - KV[e];
        }
        __syncthreads();
        mm64_inplaceA(Vv, KV);                 // V = V @ KV
        __syncthreads();
#pragma unroll
        for (int i = 0; i < 16; ++i) { int e = i * 256 + tid; Vv[e] *= 0.25f; }
        __syncthreads();
    }
#pragma unroll
    for (int i = 0; i < 16; ++i) { int e = i * 256 + tid; K2INVb[gb + e] = f2bf(Vv[e]); }
}

// ---------------------------------------------------------------------------
// Kernel 4: KV3 = softmax_rows(Ql @ K^T) @ V   [B,H,M,D], flash-style.
// One block per (b,h), 256 threads, stream N in tiles of 64.
// (logits are tiny here: |Ql.K| << 1, so exp without max-sub is safe)
// ---------------------------------------------------------------------------
__global__ __launch_bounds__(256) void k_kv3(
    const __bf16* __restrict__ Qlb, const __bf16* __restrict__ Kb,
    const __bf16* __restrict__ Vb, __bf16* __restrict__ KV3b) {
    __shared__ __align__(16) __bf16 Qls[64 * 64];
    __shared__ __align__(16) __bf16 Kt[64 * 64];
    __shared__ __align__(16) __bf16 VT[64 * 64];
    __shared__ __align__(16) __bf16 Ps[64 * 64];
    __shared__ float rowsum[64];
    const int tid  = threadIdx.x;
    const int lane = tid & 31;
    const int w8   = tid >> 5;
    const int m15  = lane & 15;
    const int lh   = lane >> 4;
    const size_t bh = blockIdx.x;

    { // stage Ql (packed u32 copies)
        const unsigned* g = (const unsigned*)Qlb + bh * 2048;
        unsigned* s = (unsigned*)Qls;
#pragma unroll
        for (int i = 0; i < 8; ++i) { int e = i * 256 + tid; s[e] = g[e]; }
    }
    if (tid < 64) rowsum[tid] = 0.f;

    const int mi  = w8 >> 1;        // output row tile (also S row tile)
    const int dj0 = (w8 & 1) * 2;   // output col tile pair / S col tile pair
    v8f acc[2];
    acc[0] = (v8f){0,0,0,0,0,0,0,0};
    acc[1] = (v8f){0,0,0,0,0,0,0,0};
    __syncthreads();

    for (int nt = 0; nt < NN / 64; ++nt) {
        const size_t tb32 = (bh * NN + (size_t)nt * 64) * (DD / 2); // u32 index
        { // K tile [n][d]
            const unsigned* g = (const unsigned*)Kb + tb32;
            unsigned* s = (unsigned*)Kt;
#pragma unroll
            for (int i = 0; i < 8; ++i) { int e = i * 256 + tid; s[e] = g[e]; }
        }
        { // V tile transposed -> VT[d][n]
            const __bf16* g = Vb + tb32 * 2;
#pragma unroll
            for (int i = 0; i < 16; ++i) {
                int e = i * 256 + tid, n = e >> 6, d = e & 63;
                VT[d * 64 + n] = g[e];
            }
        }
        __syncthreads();

        // S = Ql @ Ktile^T, exponentiate, write P (bf16)
#pragma unroll
        for (int j = 0; j < 2; ++j) {
            const int sm = mi * 16, sn = (dj0 + j) * 16;
            v8f s = (v8f){0,0,0,0,0,0,0,0};
#pragma unroll
            for (int kb = 0; kb < 64; kb += 32) {
                v16bf a = ldfrag(Qls, sm + m15, 64, kb, lane);
                v16bf b = ldfrag(Kt, sn + m15, 64, kb, lane);
                s = wmma_bf16(a, b, s);
            }
#pragma unroll
            for (int r = 0; r < 8; ++r) {
                float ev = __expf(s[r]);
                Ps[(sm + r + 8 * lh) * 64 + sn + m15] = f2bf(ev);
            }
        }
        __syncthreads();

        if (tid < 64) {                      // running denominator
            float s = rowsum[tid];
            const __bf16* pr = &Ps[tid * 64];
            for (int j = 0; j < 64; ++j) s += bf2f(pr[j]);
            rowsum[tid] = s;
        }

        // acc += P @ Vtile  (B operand from VT: col=d contiguous in n)
#pragma unroll
        for (int j = 0; j < 2; ++j) {
            const int dc = (dj0 + j) * 16;
#pragma unroll
            for (int kb = 0; kb < 64; kb += 32) {
                v16bf a = ldfrag(Ps, mi * 16 + m15, 64, kb, lane);
                v16bf b = ldfrag(VT, dc + m15, 64, kb, lane);
                acc[j] = wmma_bf16(a, b, acc[j]);
            }
        }
        __syncthreads();
    }

#pragma unroll
    for (int j = 0; j < 2; ++j)
#pragma unroll
        for (int r = 0; r < 8; ++r) {
            int m = mi * 16 + r + 8 * lh;
            int d = (dj0 + j) * 16 + m15;
            KV3b[(bh * MM + m) * DD + d] = f2bf(acc[j][r] / rowsum[m]);
        }
}

// ---------------------------------------------------------------------------
// Kernel 5: W2 = kernel2^-1 @ KV3  [B,H,M,D] bf16 (tiny, fp32 VALU)
// ---------------------------------------------------------------------------
__global__ __launch_bounds__(256) void k_w2(
    const __bf16* __restrict__ K2INVb, const __bf16* __restrict__ KV3b,
    __bf16* __restrict__ W2b) {
    __shared__ float A[64 * 64];
    __shared__ float Bm[64 * 64];
    const int tid = threadIdx.x;
    const size_t gb = (size_t)blockIdx.x * (MM * DD);
#pragma unroll
    for (int i = 0; i < 16; ++i) {
        int e = i * 256 + tid;
        A[e]  = bf2f(K2INVb[gb + e]);
        Bm[e] = bf2f(KV3b[gb + e]);
    }
    __syncthreads();
#pragma unroll
    for (int i = 0; i < 16; ++i) {
        int e = i * 256 + tid, r = e >> 6, c = e & 63;
        float a = 0.f;
#pragma unroll
        for (int k = 0; k < 64; ++k) a += A[r * 64 + k] * Bm[k * 64 + c];
        W2b[gb + e] = f2bf(a);
    }
}

// ---------------------------------------------------------------------------
// Kernel 6: SV = softmax_rows(Q @ Kl^T) @ W2 ; write [B,N,H,D] (=[B,N,C]) bf16
// grid B*H*(N/64), one block per 64 rows of one head.
// ---------------------------------------------------------------------------
__global__ __launch_bounds__(256) void k_sv(
    const __bf16* __restrict__ Qb, const __bf16* __restrict__ Klb,
    const __bf16* __restrict__ W2b, __bf16* __restrict__ SVb) {
    __shared__ __align__(16) __bf16 Qs[64 * 64];
    __shared__ __align__(16) __bf16 Kls[64 * 64];
    __shared__ __align__(16) __bf16 W2T[64 * 64];
    __shared__ __align__(16) __bf16 Ps[64 * 64];
    __shared__ float rowsum[64];
    const int tid  = threadIdx.x;
    const int lane = tid & 31;
    const int w8   = tid >> 5;
    const int m15  = lane & 15;
    const int lh   = lane >> 4;
    const int nt   = blockIdx.x & (NN / 64 - 1);
    const size_t bh = blockIdx.x >> 6;
    const int h = (int)(bh % HH), b = (int)(bh / HH);

    { // stage Q rows
        const unsigned* g = (const unsigned*)Qb + (bh * NN + (size_t)nt * 64) * (DD / 2);
        unsigned* s = (unsigned*)Qs;
#pragma unroll
        for (int i = 0; i < 8; ++i) { int e = i * 256 + tid; s[e] = g[e]; }
    }
    { // stage Kl
        const unsigned* g = (const unsigned*)Klb + bh * 2048;
        unsigned* s = (unsigned*)Kls;
#pragma unroll
        for (int i = 0; i < 8; ++i) { int e = i * 256 + tid; s[e] = g[e]; }
    }
    { // stage W2 transposed -> W2T[d][l]
        const __bf16* g = W2b + bh * (MM * DD);
#pragma unroll
        for (int i = 0; i < 16; ++i) {
            int e = i * 256 + tid, l = e >> 6, d = e & 63;
            W2T[d * 64 + l] = g[e];
        }
    }
    if (tid < 64) rowsum[tid] = 0.f;
    __syncthreads();

    const int mi  = w8 >> 1;
    const int dj0 = (w8 & 1) * 2;

    // S1 = Q @ Kl^T -> exp -> Ps  (logits tiny; no max-sub needed)
#pragma unroll
    for (int j = 0; j < 2; ++j) {
        const int sm = mi * 16, sn = (dj0 + j) * 16;
        v8f s = (v8f){0,0,0,0,0,0,0,0};
#pragma unroll
        for (int kb = 0; kb < 64; kb += 32) {
            v16bf a = ldfrag(Qs, sm + m15, 64, kb, lane);
            v16bf bop = ldfrag(Kls, sn + m15, 64, kb, lane);
            s = wmma_bf16(a, bop, s);
        }
#pragma unroll
        for (int r = 0; r < 8; ++r) {
            float ev = __expf(s[r]);
            Ps[(sm + r + 8 * lh) * 64 + sn + m15] = f2bf(ev);
        }
    }
    __syncthreads();
    if (tid < 64) {
        float s = 0.f;
        const __bf16* pr = &Ps[tid * 64];
        for (int j = 0; j < 64; ++j) s += bf2f(pr[j]);
        rowsum[tid] = s;
    }

    // SV = P @ W2
    v8f acc[2];
    acc[0] = (v8f){0,0,0,0,0,0,0,0};
    acc[1] = (v8f){0,0,0,0,0,0,0,0};
#pragma unroll
    for (int j = 0; j < 2; ++j) {
        const int dc = (dj0 + j) * 16;
#pragma unroll
        for (int kb = 0; kb < 64; kb += 32) {
            v16bf a = ldfrag(Ps, mi * 16 + m15, 64, kb, lane);
            v16bf bop = ldfrag(W2T, dc + m15, 64, kb, lane);
            acc[j] = wmma_bf16(a, bop, acc[j]);
        }
    }
    __syncthreads();

#pragma unroll
    for (int j = 0; j < 2; ++j)
#pragma unroll
        for (int r = 0; r < 8; ++r) {
            int m = mi * 16 + r + 8 * lh;
            int n = nt * 64 + m;
            int d = (dj0 + j) * 16 + m15;
            SVb[(((size_t)b * NN + n) * HH + h) * DD + d] =
                f2bf(acc[j][r] / rowsum[m]);
        }
}

// ---------------------------------------------------------------------------
// Kernel 7: out = SV @ w_proj^T + b_proj + V_residual   (fp32 out)
// grid (256, 12), block 256. Block tile 128 x 64 (one head's cols).
// ---------------------------------------------------------------------------
__global__ __launch_bounds__(256) void k_gemm_out(
    const __bf16* __restrict__ SVb, const float* __restrict__ wproj,
    const float* __restrict__ bproj, const __bf16* __restrict__ Vb,
    float* __restrict__ out) {
    __shared__ __align__(16) __bf16 As[128 * 32];
    __shared__ __align__(16) __bf16 Bs[64 * 32];
    const int tid  = threadIdx.x;
    const int lane = tid & 31;
    const int w8   = tid >> 5;
    const int wr   = w8 >> 1;
    const int wc   = w8 & 1;
    const int r0   = blockIdx.x * 128;
    const int c0   = blockIdx.y * 64;
    const int m15  = lane & 15;
    const int lh   = lane >> 4;

    v8f acc[2][2];
#pragma unroll
    for (int i = 0; i < 2; ++i)
#pragma unroll
        for (int j = 0; j < 2; ++j) acc[i][j] = (v8f){0,0,0,0,0,0,0,0};

    for (int kt = 0; kt < CC; kt += 32) {
        if (kt + 32 < CC)
            __builtin_prefetch(&SVb[(size_t)(r0 + (tid >> 1)) * CC + kt + 32], 0, 1);
        { // A: already bf16, packed u32 copy
            const unsigned* g = (const unsigned*)SVb;
            unsigned* s = (unsigned*)As;
#pragma unroll
            for (int i = 0; i < 8; ++i) {
                int e = i * 256 + tid;
                int row = e >> 4, cp = e & 15;     // 2 halves per u32
                s[e] = g[((size_t)(r0 + row) * CC + kt) / 2 + cp];
            }
        }
#pragma unroll
        for (int i = 0; i < 8; ++i) {
            int e = i * 256 + tid, row = e >> 5, col = e & 31;
            Bs[e] = f2bf(wproj[(size_t)(c0 + row) * CC + kt + col]);
        }
        __syncthreads();
#pragma unroll
        for (int i = 0; i < 2; ++i) {
            v16bf a = ldfrag(As, wr * 32 + i * 16 + m15, 32, 0, lane);
#pragma unroll
            for (int j = 0; j < 2; ++j) {
                v16bf b = ldfrag(Bs, wc * 32 + j * 16 + m15, 32, 0, lane);
                acc[i][j] = wmma_bf16(a, b, acc[i][j]);
            }
        }
        __syncthreads();
    }

    const int h = blockIdx.y;   // 64 cols == head h
#pragma unroll
    for (int i = 0; i < 2; ++i)
#pragma unroll
        for (int j = 0; j < 2; ++j)
#pragma unroll
            for (int r = 0; r < 8; ++r) {
                int m  = wr * 32 + i * 16 + r + 8 * lh;
                int gr = r0 + m;
                int b  = gr >> 12, n = gr & (NN - 1);
                int d  = wc * 32 + j * 16 + m15;
                float v = acc[i][j][r] + bproj[c0 + d] +
                          bf2f(Vb[(((size_t)b * HH + h) * NN + n) * DD + d]);
                out[(size_t)gr * CC + c0 + d] = v;
            }
}

// ---------------------------------------------------------------------------
extern "C" void kernel_launch(void* const* d_in, const int* in_sizes, int n_in,
                              void* d_out, int out_size, void* d_ws, size_t ws_size,
                              hipStream_t stream) {
    (void)in_sizes; (void)n_in; (void)out_size; (void)ws_size;
    const float* x      = (const float*)d_in[0];
    const float* w_qkv  = (const float*)d_in[1];
    const float* w_proj = (const float*)d_in[2];
    const float* b_proj = (const float*)d_in[3];
    float* out = (float*)d_out;

    // Workspace layout (bf16 halves)
    const size_t szBig = (size_t)BB * HH * NN * DD;   // 25,165,824 halves
    const size_t szSml = (size_t)BB * HH * MM * DD;   //    393,216 halves
    __bf16* Qb     = (__bf16*)d_ws;
    __bf16* Kb     = Qb + szBig;
    __bf16* Vb     = Kb + szBig;
    __bf16* SVb    = Vb + szBig;
    __bf16* Qlb    = SVb + szBig;
    __bf16* Klb    = Qlb + szSml;
    __bf16* K2INVb = Klb + szSml;
    __bf16* KV3b   = K2INVb + szSml;
    __bf16* W2b    = KV3b + szSml;

    k_gemm_qkv<<<dim3((BB * NN) / 128, (3 * CC) / 64), 256, 0, stream>>>(
        x, w_qkv, Qb, Kb, Vb);
    k_landmark<<<BB * HH * MM, 64, 0, stream>>>(Qb, Kb, Qlb, Klb);
    k_inv<<<BB * HH, 256, 0, stream>>>(Qlb, Klb, K2INVb);
    k_kv3<<<BB * HH, 256, 0, stream>>>(Qlb, Kb, Vb, KV3b);
    k_w2<<<BB * HH, 256, 0, stream>>>(K2INVb, KV3b, W2b);
    k_sv<<<BB * HH * (NN / 64), 256, 0, stream>>>(Qb, Klb, W2b, SVb);
    k_gemm_out<<<dim3((BB * NN) / 128, CC / 64), 256, 0, stream>>>(
        SVb, w_proj, b_proj, Vb, out);
}
